// FullRankRNN_91036126806946
// MI455X (gfx1250) — compile-verified, matrix operand order checked by reading, not run on hardware
//
#include <hip/hip_runtime.h>

typedef __attribute__((ext_vector_type(16))) __bf16 v16bf;
typedef __attribute__((ext_vector_type(8)))  float  v8f;

// Problem sizes (fixed by the reference)
constexpr int Bb = 128, T = 1000, I = 4, H = 512, O = 2;
constexpr float NOISE_STD = 0.05f, ALPHA = 0.2f;

constexpr int ROWS  = 16;          // batch rows per workgroup  -> 8 workgroups
constexpr int WAVES = 8;           // 256 threads, wave32
constexpr int NPW   = H / WAVES;   // 64 hidden columns per wave (4 N-tiles of 16)
constexpr int KEXT  = H + 32;      // K extended by one tile carrying the input proj
constexpr int KT    = KEXT / 32;   // 17 K-tiles of 32 (bf16 WMMA K)
constexpr int RPAD  = 552;         // padded LDS row (shorts): 1104B = 80 mod 256
                                   // -> 16 A-frag rows hit 16 disjoint bank groups

static __device__ __forceinline__ unsigned short f2bf(float x) {
  unsigned int u = __builtin_bit_cast(unsigned int, x);
  u += 0x7FFFu + ((u >> 16) & 1u);        // round-to-nearest-even
  return (unsigned short)(u >> 16);
}

// Hardware tanh (CDNA5 TRANS op). Builtin if declared, else raw v_tanh_f32
// with a trailing v_nop to satisfy the 1-op TRANS result hazard.
static __device__ __forceinline__ float htanh(float x) {
#if __has_builtin(__builtin_amdgcn_tanhf)
  return __builtin_amdgcn_tanhf(x);
#else
  float y;
  asm volatile("v_tanh_f32 %0, %1\n\tv_nop" : "=v"(y) : "v"(x));
  return y;
#endif
}

union ABFrag { uint4 q[2]; v16bf v; };

// ---------------------------------------------------------------------------
// Pre-convert [wrec ; wi*si] into bf16 WMMA B-fragment order.
// B is K x N, K = 544:  B[k][n] = wrec[n][k]          (k < 512)
//                       B[512+i][n] = wi[i][n]*si[i]  (i < 4), else 0
// Tile (kt, ntG): 32 lanes x 16 bf16 contiguous (32 B per lane).
// Lane<16: n=ntG*16+lane, k_local = (j<8 ? j : j+8); lane>=16: (j<8 ? j+8 : j+16).
// ---------------------------------------------------------------------------
__global__ __launch_bounds__(256) void convert_b_kernel(
    const float* __restrict__ wrec, const float* __restrict__ wi,
    const float* __restrict__ si, unsigned short* __restrict__ wsB) {
  int e = blockIdx.x * 256 + threadIdx.x;
  if (e >= KT * 32 * 512) return;
  int tile   = e >> 9;          // 512 elements per tile
  int within = e & 511;
  int lane   = within >> 4;
  int j      = within & 15;
  int kt     = tile >> 5;       // 0..16
  int ntG    = tile & 31;       // 0..31
  int n      = ntG * 16 + (lane & 15);
  int kl     = (lane < 16) ? ((j < 8) ? j : j + 8)
                           : ((j < 8) ? j + 8 : j + 16);
  float val;
  if (kt < 16) {
    val = wrec[n * H + (kt * 32 + kl)];
  } else {
    val = (kl < I) ? wi[kl * H + n] * si[kl] : 0.f;
  }
  wsB[e] = f2bf(val);
}

// ---------------------------------------------------------------------------
// Main scan kernel: one workgroup owns 16 batch rows for all 1000 steps.
// Only 8 blocks x 8 waves ever run -> at most 2 waves/SIMD.
// ---------------------------------------------------------------------------
__global__ __launch_bounds__(256)
__attribute__((amdgpu_waves_per_eu(2)))
void rnn_scan_kernel(
    const float* __restrict__ input, const float* __restrict__ noise,
    const float* __restrict__ wo,    const float* __restrict__ so,
    const float* __restrict__ h0,    float* __restrict__ out,
    const unsigned short* __restrict__ wsB) {

  __shared__ __align__(16) unsigned short rbuf[2][ROWS][RPAD]; // [r | inp-ext], bf16
  __shared__ float outSlots[2][WAVES][ROWS][O];                // per-wave partials

  const int tid     = threadIdx.x;
  const int lane    = tid & 31;
  const int w       = tid >> 5;
  const int rowBase = blockIdx.x * ROWS;

  const int nCol = lane & 15;
  const int rOff = (lane < 16) ? 0 : 8;   // C/D row offset per ISA layout
  const int sA   = (lane < 16) ? 0 : 8;   // A-frag K sub-offset (elements)
  const int mA   = lane & 15;             // A-frag row

  // ---- one-time staging ----
  // r(0) = tanh(h0), broadcast over rows
  for (int idx = tid; idx < ROWS * H; idx += 256) {
    int m = idx >> 9, n = idx & (H - 1);
    rbuf[0][m][n] = f2bf(htanh(h0[n]));
  }
  // zero the K-extension + pad columns of both buffers
  for (int idx = tid; idx < 2 * ROWS * (RPAD - H); idx += 256) {
    int b = idx / (ROWS * (RPAD - H));
    int r = idx % (ROWS * (RPAD - H));
    rbuf[b][r / (RPAD - H)][H + r % (RPAD - H)] = 0;
  }
  __syncthreads();   // zeros visible before the t=0 input-ext write below
  if (tid < ROWS) {  // input extension for t = 0
    const float4* ip = (const float4*)input;   // I == 4 floats per (b,t)
    float4 v = ip[(size_t)(rowBase + tid) * T + 0];
    rbuf[0][tid][H + 0] = f2bf(v.x); rbuf[0][tid][H + 1] = f2bf(v.y);
    rbuf[0][tid][H + 2] = f2bf(v.z); rbuf[0][tid][H + 3] = f2bf(v.w);
  }

  // loop-invariant per-lane values
  float h[4][8], wov0[4], wov1[4];
  #pragma unroll
  for (int nt = 0; nt < 4; ++nt) {
    int n = w * NPW + nt * 16 + nCol;
    float h0n = h0[n];
    #pragma unroll
    for (int v = 0; v < 8; ++v) h[nt][v] = h0n;
    wov0[nt] = wo[n * O + 0] * so[0];
    wov1[nt] = wo[n * O + 1] * so[1];
  }
  const float* nptr[8];
  #pragma unroll
  for (int v = 0; v < 8; ++v)
    nptr[v] = noise + (size_t)(rowBase + rOff + v) * ((size_t)T * H)
                    + w * NPW + nCol;

  __syncthreads();

  int cur = 0;
  for (int t = 0; t < T; ++t) {
    const unsigned short* __restrict__ rc = &rbuf[cur][0][0];
    unsigned short* __restrict__ rn = &rbuf[cur ^ 1][0][0];

    // Defeat LICM on the (loop-invariant) B-matrix loads without losing the
    // global address space: keep wb a GEP on wsB with a laundered zero offset
    // so the loads stay inside the t-loop as global_load_b128.
    unsigned long long delta = 0;
    asm volatile("" : "+s"(delta));
    const unsigned short* wb = wsB + delta;

    // ---- prefetch this step's noise (non-temporal: keep wsB hot in L2) ----
    float nz[4][8];
    #pragma unroll
    for (int v = 0; v < 8; ++v)
      #pragma unroll
      for (int nt = 0; nt < 4; ++nt)
        nz[nt][v] = __builtin_nontemporal_load(nptr[v] + nt * 16);

    // ---- GEMM: acc = [r | inp] (16x544) @ [wrecT ; wi_full] (544x64) ----
    v8f acc[4] = {v8f{}, v8f{}, v8f{}, v8f{}};
    const unsigned short* rrow = rc + mA * RPAD;
    ABFrag a3[3], b3[3][4];

    auto loadGroup = [&](int g, int slot) {
      a3[slot].q[0] = *(const uint4*)(rrow + g * 32 + sA);
      a3[slot].q[1] = *(const uint4*)(rrow + g * 32 + sA + 16);
      const unsigned short* bb =
          wb + ((size_t)(g * 32 + w * 4) * 32 + lane) * 16;
      #pragma unroll
      for (int nt = 0; nt < 4; ++nt) {
        b3[slot][nt].q[0] = *(const uint4*)(bb + nt * 512);
        b3[slot][nt].q[1] = *(const uint4*)(bb + nt * 512 + 8);
      }
    };

    loadGroup(0, 0);
    loadGroup(1, 1);
    #pragma unroll
    for (int kt = 0; kt < KT; ++kt) {
      const int s = kt % 3;
      if (kt + 2 < KT) loadGroup(kt + 2, (kt + 2) % 3);
      #pragma unroll
      for (int nt = 0; nt < 4; ++nt)
        acc[nt] = __builtin_amdgcn_wmma_f32_16x16x32_bf16(
            false, a3[s].v, false, b3[s][nt].v, (short)0, acc[nt],
            false, false);
    }

#if __has_builtin(__builtin_amdgcn_sched_group_barrier)
    // Pin the region schedule into a 2-deep software pipeline
    // (masks: 0x008 WMMA, 0x020 VMEM read, 0x100 DS read):
    //   [32 noise] [A0+B0] [A1+B1] { [WMMA g] [A(g+2)+B(g+2)] }x15 [WMMA]x2
    __builtin_amdgcn_sched_group_barrier(0x020, 32, 0);   // noise prefetch
    __builtin_amdgcn_sched_group_barrier(0x100, 2, 0);    // A group 0 (LDS)
    __builtin_amdgcn_sched_group_barrier(0x020, 8, 0);    // B group 0 (L2)
    __builtin_amdgcn_sched_group_barrier(0x100, 2, 0);    // A group 1
    __builtin_amdgcn_sched_group_barrier(0x020, 8, 0);    // B group 1
    #pragma unroll
    for (int g = 0; g < KT - 2; ++g) {
      __builtin_amdgcn_sched_group_barrier(0x008, 4, 0);  // WMMA group g
      __builtin_amdgcn_sched_group_barrier(0x100, 2, 0);  // A group g+2
      __builtin_amdgcn_sched_group_barrier(0x020, 8, 0);  // B group g+2
    }
    __builtin_amdgcn_sched_group_barrier(0x008, 4, 0);    // WMMA group KT-2
    __builtin_amdgcn_sched_group_barrier(0x008, 4, 0);    // WMMA group KT-1
#endif

    // ---- stage input extension for t+1 into the next buffer ----
    if (tid < ROWS && t + 1 < T) {
      const float4* ip = (const float4*)input;
      float4 v = ip[(size_t)(rowBase + tid) * T + (t + 1)];
      unsigned short* e = rn + tid * RPAD + H;
      e[0] = f2bf(v.x); e[1] = f2bf(v.y); e[2] = f2bf(v.z); e[3] = f2bf(v.w);
    }

    // ---- fused epilogue: leaky update, hw tanh, r store, out partials ----
    float po[8][2];
    #pragma unroll
    for (int v = 0; v < 8; ++v) { po[v][0] = 0.f; po[v][1] = 0.f; }

    #pragma unroll
    for (int nt = 0; nt < 4; ++nt) {
      const int n = w * NPW + nt * 16 + nCol;
      #pragma unroll
      for (int v = 0; v < 8; ++v) {
        const int m = rOff + v;
        float hn = (1.f - ALPHA) * h[nt][v] + ALPHA * acc[nt][v]
                 + NOISE_STD * nz[nt][v];
        h[nt][v] = hn;
        float r = htanh(hn);
        rn[m * RPAD + n] = f2bf(r);
        po[v][0] += r * wov0[nt];
        po[v][1] += r * wov1[nt];
      }
    }
    #pragma unroll
    for (int v = 0; v < 8; ++v) nptr[v] += H;

    // reduce output partials across the 16-lane half (element-wise butterfly)
    #pragma unroll
    for (int mask = 1; mask < 16; mask <<= 1) {
      #pragma unroll
      for (int v = 0; v < 8; ++v) {
        po[v][0] += __shfl_xor(po[v][0], mask, 32);
        po[v][1] += __shfl_xor(po[v][1], mask, 32);
      }
    }
    if (nCol == 0) {          // lanes 0 (rows 0-7) and 16 (rows 8-15)
      #pragma unroll
      for (int v = 0; v < 8; ++v) {
        outSlots[cur][w][rOff + v][0] = po[v][0];
        outSlots[cur][w][rOff + v][1] = po[v][1];
      }
    }

    __syncthreads();   // r(t+1), inp-ext(t+1), outSlots(t) all complete

    if (tid < ROWS * O) {
      int m = tid >> 1, o = tid & 1;
      float s = 0.f;
      #pragma unroll
      for (int ww = 0; ww < WAVES; ++ww) s += outSlots[cur][ww][m][o];
      __builtin_nontemporal_store(
          s, &out[(size_t)(rowBase + m) * ((size_t)T * O) + (size_t)t * O + o]);
    }
    cur ^= 1;
  }
}

// ---------------------------------------------------------------------------
extern "C" void kernel_launch(void* const* d_in, const int* in_sizes, int n_in,
                              void* d_out, int out_size, void* d_ws, size_t ws_size,
                              hipStream_t stream) {
  (void)in_sizes; (void)n_in; (void)out_size; (void)ws_size;
  const float* input = (const float*)d_in[0];
  const float* noise = (const float*)d_in[1];
  const float* wi    = (const float*)d_in[2];
  const float* si    = (const float*)d_in[3];
  const float* wrec  = (const float*)d_in[4];
  const float* wo    = (const float*)d_in[5];
  const float* so    = (const float*)d_in[6];
  const float* h0    = (const float*)d_in[7];
  float* out = (float*)d_out;
  unsigned short* wsB = (unsigned short*)d_ws;   // KT*32*512 bf16 = 544 KB scratch

  const int NB = KT * 32 * 512;
  convert_b_kernel<<<(NB + 255) / 256, 256, 0, stream>>>(wrec, wi, si, wsB);
  rnn_scan_kernel<<<Bb / ROWS, 256, 0, stream>>>(input, noise, wo, so,
                                                 h0, out, wsB);
}
